// DecoderStack_82162724373197
// MI455X (gfx1250) — compile-verified, matrix operand order checked by reading, not run on hardware
//
#include <hip/hip_runtime.h>
#include <hip/hip_bf16.h>
#include <math.h>
#include <stdint.h>

#define Bn   2
#define Fq   512
#define Tn   512
#define Dm   1024
#define Nh   16
#define Hd   64
#define Kt   32
#define FILTN 4096
#define Ln   4
#define EPSV 1e-6f

typedef __bf16 bf16;
typedef __attribute__((ext_vector_type(8)))  bf16  v8bf;
typedef __attribute__((ext_vector_type(16))) bf16  v16bf;
typedef __attribute__((ext_vector_type(8)))  float v8f;
typedef __attribute__((ext_vector_type(4)))  int   v4i;

// ---- CDNA5 async global->LDS copy (ASYNCcnt path), with portable fallback ---
#if __has_builtin(__builtin_amdgcn_global_load_async_to_lds_b128) && \
    __has_builtin(__builtin_amdgcn_s_wait_asynccnt)
#define HAVE_ASYNC 1
typedef __attribute__((address_space(1))) v4i as1_v4i;  // global
typedef __attribute__((address_space(3))) v4i as3_v4i;  // LDS
#endif

__device__ __forceinline__ void async_cp16(const void* g, void* l) {
#ifdef HAVE_ASYNC
    __builtin_amdgcn_global_load_async_to_lds_b128(
        (as1_v4i*)(uintptr_t)g, (as3_v4i*)(uintptr_t)l, 0, 0);
#else
    *(v8bf*)l = *(const v8bf*)g;
#endif
}

__device__ __forceinline__ void async_wait() {
#ifdef HAVE_ASYNC
    __builtin_amdgcn_s_wait_asynccnt(0);
#endif
}

// ---- WMMA helpers -----------------------------------------------------------
// A-frag (16x32 bf16, row-major source in LDS with given stride):
// lane<16: row M=lane,     K chunks {0..7, 16..23}
// lane>=16: row M=lane-16, K chunks {8..15, 24..31}
// B-frag is symmetric if the LDS tile is stored [N][K] (i.e. transposed).
__device__ __forceinline__ v16bf frag_ld(const bf16* base, int row0, int stride, int lane) {
    int hv = (lane >> 4) & 1;
    const bf16* p = base + (size_t)(row0 + (lane & 15)) * stride + hv * 8;
    v8bf lo = *(const v8bf*)(p);
    v8bf hh = *(const v8bf*)(p + 16);
    v16bf f;
#pragma unroll
    for (int i = 0; i < 8; ++i) { f[i] = lo[i]; f[i + 8] = hh[i]; }
    return f;
}

__device__ __forceinline__ v8f wmma_bf16(v16bf a, v16bf b, v8f c) {
    return __builtin_amdgcn_wmma_f32_16x16x32_bf16(false, a, false, b, (short)0, c, false, false);
}

// ---- Generic GEMM: Out[M,Nd] = epi(A[M,Kd] @ W[Kd,Nd]) ----------------------
// flags: 1=relu, 2=+bias[col], 4=+resid[row,col], 8=write bf16 to OutH
// (instead of fp32 Out); scale applied to acc first. GK=64: 8 WMMAs/wave/barrier.
#define GM 64
#define GN 128
#define GK 64
#define LDA 72
#define LDB 72

__global__ void __launch_bounds__(256)
gemm_bf16_kernel(const float* __restrict__ A, const float* __restrict__ W,
                 const float* __restrict__ bias, const float* __restrict__ resid,
                 float* __restrict__ Out, bf16* __restrict__ OutH,
                 int M, int Nd, int Kd, float scale, int flags)
{
    __shared__ bf16 sA[GM * LDA];
    __shared__ bf16 sB[GN * LDB];
    int tid  = threadIdx.x;
    int lane = tid & 31, wid = tid >> 5;
    int m0 = blockIdx.y * GM;
    int n0 = blockIdx.x * GN;
    int wm = (wid & 1) * 32;
    int wn = (wid >> 1) * 32;

    v8f acc[2][2];
#pragma unroll
    for (int i = 0; i < 2; ++i)
#pragma unroll
        for (int j = 0; j < 2; ++j)
#pragma unroll
            for (int e = 0; e < 8; ++e) acc[i][j][e] = 0.f;

    for (int k0 = 0; k0 < Kd; k0 += GK) {
        // stage A tile (fp32 -> bf16)
        for (int i = tid; i < GM * GK; i += 256) {
            int k = i & 63, m = i >> 6;
            sA[m * LDA + k] = (bf16)A[(size_t)(m0 + m) * Kd + k0 + k];
        }
        // stage W tile transposed: sB[n][k]
        for (int i = tid; i < GK * GN; i += 256) {
            int n = i & 127, k = i >> 7;
            sB[n * LDB + k] = (bf16)W[(size_t)(k0 + k) * Nd + n0 + n];
        }
        if (k0 + GK < Kd)
            __builtin_prefetch(&W[(size_t)(k0 + GK) * Nd + n0 + (tid & 127)], 0, 1);
        __syncthreads();

#pragma unroll
        for (int ks = 0; ks < GK; ks += 32) {
            v16bf a0 = frag_ld(sA + ks, wm,      LDA, lane);
            v16bf a1 = frag_ld(sA + ks, wm + 16, LDA, lane);
            v16bf b0 = frag_ld(sB + ks, wn,      LDB, lane);
            v16bf b1 = frag_ld(sB + ks, wn + 16, LDB, lane);
            acc[0][0] = wmma_bf16(a0, b0, acc[0][0]);
            acc[0][1] = wmma_bf16(a0, b1, acc[0][1]);
            acc[1][0] = wmma_bf16(a1, b0, acc[1][0]);
            acc[1][1] = wmma_bf16(a1, b1, acc[1][1]);
        }
        __syncthreads();
    }

    int hv = lane >> 4, nc = lane & 15;
#pragma unroll
    for (int i = 0; i < 2; ++i)
#pragma unroll
        for (int j = 0; j < 2; ++j) {
            int colg = n0 + wn + j * 16 + nc;
            float bv = (flags & 2) ? bias[colg] : 0.f;
#pragma unroll
            for (int r = 0; r < 8; ++r) {
                int rowg = m0 + wm + i * 16 + r + hv * 8;
                float v = acc[i][j][r] * scale + bv;
                if (flags & 1) v = fmaxf(v, 0.f);
                if (flags & 4) v += resid[(size_t)rowg * Nd + colg];
                if (flags & 8) OutH[(size_t)rowg * Nd + colg] = (bf16)v;
                else           Out [(size_t)rowg * Nd + colg] = v;
            }
        }
}

// ---- Row LayerNorm (non-trainable): (x-mean)/(std+eps), one wave per row ----
__global__ void __launch_bounds__(256)
norm_rows_kernel(const float* __restrict__ in, float* __restrict__ out, int cols)
{
    int lane = threadIdx.x & 31;
    int row  = blockIdx.x * 8 + (threadIdx.x >> 5);
    const float* p = in + (size_t)row * cols;
    float s = 0.f;
    for (int c = lane; c < cols; c += 32) s += p[c];
#pragma unroll
    for (int o = 16; o > 0; o >>= 1) s += __shfl_xor(s, o, 32);
    float mean = s / cols;
    float s2 = 0.f;
    for (int c = lane; c < cols; c += 32) { float d = p[c] - mean; s2 += d * d; }
#pragma unroll
    for (int o = 16; o > 0; o >>= 1) s2 += __shfl_xor(s2, o, 32);
    float inv = 1.f / (sqrtf(s2 / cols) + EPSV);
    float* q = out + (size_t)row * cols;
    for (int c = lane; c < cols; c += 32) q[c] = (p[c] - mean) * inv;
}

// ---- Time-bias MLP: qs[l,b,f,t] = relu(d*Wth+bth) @ Wto + bto ---------------
__global__ void __launch_bounds__(256)
time_bias_kernel(const float* __restrict__ dist, const float* __restrict__ Wth,
                 const float* __restrict__ bth, const float* __restrict__ Wto,
                 const float* __restrict__ bto, float* __restrict__ qs)
{
    int idx = blockIdx.x * 256 + threadIdx.x;
    int per = Bn * Fq * Tn;
    if (idx >= Ln * per) return;
    int l = idx / per;
    int r = idx - l * per;
    float d = dist[r];
    const float* wth  = Wth + l * Kt;
    const float* bthl = bth + l * Kt;
    const float* wto  = Wto + l * Kt;
    float a = bto[l];
#pragma unroll
    for (int k = 0; k < Kt; ++k)
        a += fmaxf(fmaf(d, wth[k], bthl[k]), 0.f) * wto[k];
    qs[idx] = a;
}

// ---- Attention: per (b, head, 64-query tile). T=512 kept fully in LDS -------
// q/k/v arrive as bf16 (written by the projection GEMMs), so Q/K staging is a
// raw 16B-granular copy -> GLOBAL_LOAD_ASYNC_TO_LDS_B128 when available.
#define FT  64
#define LQK 72    // bf16 stride for 64-wide rows (144B, 16B aligned)
#define LSS 516   // f32 stride for 512-wide rows
#define LPP 520   // bf16 stride for 512-wide rows
#define ATT_LDS (FT*LQK*2 + Tn*LQK*2 + FT*LSS*4 + FT*LPP*2)  // 281600 B

__global__ void __launch_bounds__(256)
attention_kernel(const bf16* __restrict__ q, const bf16* __restrict__ k,
                 const bf16* __restrict__ v, const float* __restrict__ qs,
                 const float* __restrict__ encb, float* __restrict__ out)
{
    extern __shared__ char smem[];
    bf16*  sQ  = (bf16*)smem;                                   // [64][LQK]
    bf16*  sKV = (bf16*)(smem + FT * LQK * 2);                  // [512][LQK] / [64][LPP]
    float* sS  = (float*)(smem + FT * LQK * 2 + Tn * LQK * 2);  // [64][LSS]
    bf16*  sP  = (bf16*)((char*)sS + FT * LSS * 4);             // [64][LPP]

    int tid = threadIdx.x;
    int lane = tid & 31, wid = tid >> 5;
    int bidx = blockIdx.x;
    int fblk = bidx & 7;
    int n    = (bidx >> 3) & 15;
    int b    = bidx >> 7;
    int f0   = fblk * FT;

    // stage Q tile and full K via 16B async copies; K kept [t][h] (B-frag layout)
    for (int i = tid; i < FT * 8; i += 256) {
        int f = i >> 3, c = i & 7;
        async_cp16(q + ((size_t)(b * Fq + f0 + f) * Nh + n) * Hd + c * 8,
                   sQ + f * LQK + c * 8);
    }
    for (int i = tid; i < Tn * 8; i += 256) {
        int t = i >> 3, c = i & 7;
        async_cp16(k + ((size_t)(b * Tn + t) * Nh + n) * Hd + c * 8,
                   sKV + t * LQK + c * 8);
    }
    async_wait();
    __syncthreads();

    // Phase 1: S = Q . K^T  (each wave owns a 64-wide t stripe)
    int tbase = wid * 64;
#pragma unroll
    for (int ft = 0; ft < 4; ++ft) {
        v16bf a0 = frag_ld(sQ,      ft * 16, LQK, lane);
        v16bf a1 = frag_ld(sQ + 32, ft * 16, LQK, lane);
#pragma unroll
        for (int tt = 0; tt < 4; ++tt) {
            v8f acc;
#pragma unroll
            for (int e = 0; e < 8; ++e) acc[e] = 0.f;
            v16bf b0 = frag_ld(sKV,      tbase + tt * 16, LQK, lane);
            v16bf b1 = frag_ld(sKV + 32, tbase + tt * 16, LQK, lane);
            acc = wmma_bf16(a0, b0, acc);
            acc = wmma_bf16(a1, b1, acc);
            int hv = lane >> 4, nc = lane & 15;
#pragma unroll
            for (int r = 0; r < 8; ++r)
                sS[(ft * 16 + r + hv * 8) * LSS + tbase + tt * 16 + nc] = acc[r];
        }
    }
    __syncthreads();

    // Phase 2: softmax over T, + time bias + enc-dec bias; write bf16 P
    for (int rr = 0; rr < 8; ++rr) {
        int fl = wid * 8 + rr;
        const float* qrow = qs + (size_t)(b * Fq + f0 + fl) * Tn;
        float vals[16];
        float mx = -3.4e38f;
#pragma unroll
        for (int i2 = 0; i2 < 16; ++i2) {
            int t = lane + i2 * 32;
            float s = sS[fl * LSS + t] + qrow[t] + encb[b * Tn + t];
            vals[i2] = s;
            mx = fmaxf(mx, s);
        }
#pragma unroll
        for (int o = 16; o > 0; o >>= 1) mx = fmaxf(mx, __shfl_xor(mx, o, 32));
        float sum = 0.f;
#pragma unroll
        for (int i2 = 0; i2 < 16; ++i2) { vals[i2] = __expf(vals[i2] - mx); sum += vals[i2]; }
#pragma unroll
        for (int o = 16; o > 0; o >>= 1) sum += __shfl_xor(sum, o, 32);
        float inv = 1.f / sum;
#pragma unroll
        for (int i2 = 0; i2 < 16; ++i2)
            sP[fl * LPP + lane + i2 * 32] = (bf16)(vals[i2] * inv);
    }
    __syncthreads();

    // Phase 3: stage V transposed [h][t] over sKV (bf16->bf16), then O = P @ V
    for (int i = tid; i < Tn * Hd; i += 256) {
        int h = i & 63, t = i >> 6;
        sKV[h * LPP + t] = v[((size_t)(b * Tn + t) * Nh + n) * Hd + h];
    }
    __syncthreads();

#pragma unroll
    for (int j = 0; j < 2; ++j) {
        int tileid = wid * 2 + j;
        int fo = tileid >> 2, ho = tileid & 3;
        v8f acc;
#pragma unroll
        for (int e = 0; e < 8; ++e) acc[e] = 0.f;
#pragma unroll
        for (int ks = 0; ks < 16; ++ks) {
            v16bf a  = frag_ld(sP  + ks * 32, fo * 16, LPP, lane);
            v16bf bb = frag_ld(sKV + ks * 32, ho * 16, LPP, lane);
            acc = wmma_bf16(a, bb, acc);
        }
        int hv = lane >> 4, nc = lane & 15;
#pragma unroll
        for (int r = 0; r < 8; ++r) {
            int f = f0 + fo * 16 + r + hv * 8;
            out[((size_t)(b * Fq + f) * Nh + n) * Hd + ho * 16 + nc] = acc[r];
        }
    }
}

// ---- Host orchestration -----------------------------------------------------
extern "C" void kernel_launch(void* const* d_in, const int* in_sizes, int n_in,
                              void* d_out, int out_size, void* d_ws, size_t ws_size,
                              hipStream_t stream)
{
    (void)in_sizes; (void)n_in; (void)out_size; (void)ws_size;
    const float* dec_in = (const float*)d_in[0];
    const float* enc    = (const float*)d_in[1];
    const float* dist   = (const float*)d_in[3];   // decoder_encoder_times_dist
    const float* encb   = (const float*)d_in[5];   // enc_dec_attn_bias
    const float* Wq  = (const float*)d_in[6];
    const float* Wk  = (const float*)d_in[7];
    const float* Wv  = (const float*)d_in[8];
    const float* Wo  = (const float*)d_in[9];
    const float* Wth = (const float*)d_in[10];
    const float* bth = (const float*)d_in[11];
    const float* Wto = (const float*)d_in[12];
    const float* bto = (const float*)d_in[13];
    const float* Wf1 = (const float*)d_in[14];
    const float* bf1 = (const float*)d_in[15];
    const float* Wf2 = (const float*)d_in[16];
    const float* bf2 = (const float*)d_in[17];

    const size_t BFD = (size_t)Bn * Fq * Dm;   // 1,048,576
    char* ws = (char*)d_ws;
    size_t off = 0;
    float* x      = (float*)(ws + off); off += BFD * 4;
    float* xn     = (float*)(ws + off); off += BFD * 4;
    float* ao     = (float*)(ws + off); off += BFD * 4;
    float* hbuf   = (float*)(ws + off); off += (size_t)Bn * Fq * FILTN * 4;
    float* qs_all = (float*)(ws + off); off += (size_t)Ln * Bn * Fq * Tn * 4;
    bf16*  qh     = (bf16*)(ws + off);  off += BFD * 2;
    bf16*  kh     = (bf16*)(ws + off);  off += BFD * 2;
    bf16*  vh     = (bf16*)(ws + off);  off += BFD * 2;

    (void)hipMemcpyAsync(x, dec_in, BFD * 4, hipMemcpyDeviceToDevice, stream);

    {
        int total = Ln * Bn * Fq * Tn;
        time_bias_kernel<<<(total + 255) / 256, 256, 0, stream>>>(dist, Wth, bth, Wto, bto, qs_all);
    }

    const int M = Bn * Fq;                // 1024 (== Bn*Tn as well)
    const float scaleQ = 0.125f;          // H^-0.5 = 64^-0.5
    dim3 gQKV(Dm / GN, M / GM);
    dim3 gF1(FILTN / GN, M / GM);

    for (int l = 0; l < Ln; ++l) {
        const float* Wq_l  = Wq  + (size_t)l * Dm * Nh * Hd;
        const float* Wk_l  = Wk  + (size_t)l * Dm * Nh * Hd;
        const float* Wv_l  = Wv  + (size_t)l * Dm * Nh * Hd;
        const float* Wo_l  = Wo  + (size_t)l * Nh * Hd * Dm;
        const float* Wf1_l = Wf1 + (size_t)l * Dm * FILTN;
        const float* bf1_l = bf1 + (size_t)l * FILTN;
        const float* Wf2_l = Wf2 + (size_t)l * FILTN * Dm;
        const float* bf2_l = bf2 + (size_t)l * Dm;
        const float* qs_l  = qs_all + (size_t)l * Bn * Fq * Tn;

        // attention sublayer (pre-norm, residual folded into O-proj epilogue)
        norm_rows_kernel<<<M / 8, 256, 0, stream>>>(x, xn, Dm);
        gemm_bf16_kernel<<<gQKV, 256, 0, stream>>>(xn,  Wq_l, nullptr, nullptr, nullptr, qh, M, Dm, Dm, scaleQ, 8);
        gemm_bf16_kernel<<<gQKV, 256, 0, stream>>>(enc, Wk_l, nullptr, nullptr, nullptr, kh, M, Dm, Dm, 1.f, 8);
        gemm_bf16_kernel<<<gQKV, 256, 0, stream>>>(enc, Wv_l, nullptr, nullptr, nullptr, vh, M, Dm, Dm, 1.f, 8);
        attention_kernel<<<Bn * Nh * (Fq / FT), 256, ATT_LDS, stream>>>(qh, kh, vh, qs_l, encb, ao);
        gemm_bf16_kernel<<<gQKV, 256, 0, stream>>>(ao, Wo_l, nullptr, x, x, nullptr, M, Dm, Dm, 1.f, 4);

        // FFN sublayer
        norm_rows_kernel<<<M / 8, 256, 0, stream>>>(x, xn, Dm);
        gemm_bf16_kernel<<<gF1, 256, 0, stream>>>(xn,   Wf1_l, bf1_l, nullptr, hbuf, nullptr, M, FILTN, Dm, 1.f, 1 | 2);
        gemm_bf16_kernel<<<gQKV, 256, 0, stream>>>(hbuf, Wf2_l, bf2_l, x,       x,   nullptr, M, Dm, FILTN, 1.f, 2 | 4);
    }

    norm_rows_kernel<<<M / 8, 256, 0, stream>>>(x, (float*)d_out, Dm);
}
